// AirFormer_79508434583691
// MI455X (gfx1250) — compile-verified, matrix operand order checked by reading, not run on hardware
//
#include <hip/hip_runtime.h>

// =====================================================================
// AirFormer forward for MI455X (gfx1250), wave32 + WMMA bf16 + TDM.
//
//  Generic WMMA GEMM:
//    block = 256 threads (8 waves); block tile 128(M) x 64(N), K step 32.
//    Each wave computes a 32x32 sub-tile = 2x2 v_wmma_f32_16x16x32_bf16,
//    f32 accumulation (4 WMMAs per K step).
//    A tile (the HBM-streaming operand) is staged with the Tensor Data
//    Mover: tensor_load_to_lds with a 2D D# (32x128 tile, tensor dims set
//    to the remaining M/K extent so hardware zero-fills the tails), LDS
//    row padded 32->36 DWORDs via the D# pad fields. Wave 0 issues the
//    DMA + s_wait_tensorcnt 0, then a block barrier publishes the tile.
//    Falls back to vectorized VALU staging when the TDM builtin is absent.
//    B tile is staged transposed into bf16 LDS (Bs[n][k]) so fragments
//    are two contiguous ds_load_b128 per 16x16 operand.
//  Elementwise stages (PE, gathers, LN, softmax, readout) are
//  bandwidth-bound VALU kernels.
// =====================================================================

#define AF_N      65536
#define AF_M      4096
#define AF_KAGG   16
#define AF_KQ     8
#define AF_DLOC   128
#define AF_DANC   256
#define AF_L      12
#define AF_H      16
#define AF_DH     16
#define AF_G      128
#define AF_TOUT   5
#define AF_INDIM  66   // 15 feats + 3 pos + 24 sin + 24 cos
#define AF_INPAD  68   // padded to a multiple of 4 for vector loads

#define BM 128
#define BN 64
#define BK 32
#define LDSP 40        // bf16 row stride (80B, 16B aligned, conflict-free)
#define AFP  36        // f32 A-tile row stride for TDM path (144B)

#if defined(__AMDGCN__) && __has_builtin(__builtin_amdgcn_tensor_load_to_lds) && \
    __has_builtin(__builtin_amdgcn_s_wait_tensorcnt)
#define USE_TDM 1
#else
#define USE_TDM 0
#endif

typedef __attribute__((ext_vector_type(4)))  float    fx4;
typedef __attribute__((ext_vector_type(4)))  __bf16   bf16x4;
typedef __attribute__((ext_vector_type(8)))  __bf16   bf16x8;
typedef __attribute__((ext_vector_type(16))) __bf16   bf16x16;
typedef __attribute__((ext_vector_type(8)))  float    floatx8;
typedef __attribute__((ext_vector_type(4)))  unsigned v4u;
typedef __attribute__((ext_vector_type(8)))  int      v8i;
typedef __attribute__((ext_vector_type(4)))  int      v4i;

__device__ __forceinline__ float af_gelu(float x) {
    return 0.5f * x * (1.0f + erff(x * 0.70710678118654752f));
}

__device__ __forceinline__ bf16x8 cvt8(fx4 a, fx4 b) {
    bf16x4 lo = __builtin_convertvector(a, bf16x4);
    bf16x4 hi = __builtin_convertvector(b, bf16x4);
    return __builtin_shufflevector(lo, hi, 0, 1, 2, 3, 4, 5, 6, 7);
}

// ---------------------------------------------------------------------
// Generic batched WMMA GEMM: C[b] = act((opA(A[b]) @ opB(B[b]))*scale
//                                        + bias[b]) + res[b]
// ---------------------------------------------------------------------
__global__ __launch_bounds__(256)
void wmma_gemm(const float* __restrict__ A, int lda, long long sA, int transA,
               const float* __restrict__ B, int ldb, long long sB, int transB,
               const float* __restrict__ bias, long long sBias,
               const float* __restrict__ res, long long sRes, int ldres,
               float* __restrict__ C, int ldc, long long sC,
               int Mr, int Nc, int Kd, float scale, int act) {
#if USE_TDM
    __shared__ float Af[BM][AFP];      // f32 A tile, DMA'd by the TDM
#else
    __shared__ __bf16 As[BM][LDSP];
#endif
    __shared__ __bf16 Bs[BN][LDSP];

    const int bm  = blockIdx.x * BM;
    const int bn  = blockIdx.y * BN;
    const int bat = blockIdx.z;

    const float* Ab = A + (long long)bat * sA;
    const float* Bb = B + (long long)bat * sB;

    const int tid  = threadIdx.x;
    const int wave = tid >> 5;
    const int lane = tid & 31;
    const int wm   = wave >> 1;       // 0..3 -> M sub-tile (32 rows)
    const int wn   = wave & 1;        // 0..1 -> N sub-tile (32 cols)
    const int g    = lane >> 4;       // K-half selector
    const int lr   = lane & 15;

    floatx8 acc[2][2] = {};

    const bool ldaAligned = ((lda & 3) == 0);
    const bool ldbAligned = ((ldb & 3) == 0);
    (void)ldaAligned;

    for (int k0 = 0; k0 < Kd; k0 += BK) {
        const bool fullK = (k0 + BK <= Kd);

        // ---------- stage A (BM x BK) ----------
#if USE_TDM
        if (!transA) {
            if (wave == 0) {
                // Tensor DMA descriptor (D#): 2D tile BKxBM, f32 elements,
                // hardware zero-fill beyond (Kd-k0, Mr-bm), LDS row padded
                // by 4 DWORDs per 32-DWORD row -> 144B stride (== AFP).
                unsigned long long ga =
                    (unsigned long long)(Ab + (long long)bm * lda + k0);
                unsigned ldsb = (unsigned)(unsigned long long)&Af[0][0];
                v4u g0 = {};
                g0[0] = 1u;                                   // count=1 (valid)
                g0[1] = ldsb;                                 // lds_addr
                g0[2] = (unsigned)(ga & 0xFFFFFFFFull);       // global_addr lo
                g0[3] = (unsigned)((ga >> 32) & 0x01FFFFFFull) | (2u << 30); // hi|type=2
                unsigned td0 = (unsigned)(Kd - k0);           // tensor extent K
                unsigned td1 = (unsigned)(Mr - bm);           // tensor extent M
                v8i g1 = {};
                g1[0] = (int)((2u << 16)      // data_size = 4B
                            | (1u << 20)      // pad_enable
                            | (4u << 22)      // pad_interval = 32 DWORDs
                            | (3u << 25));    // pad_amount   = 4 DWORDs
                g1[1] = (int)((td0 & 0xFFFFu) << 16);
                g1[2] = (int)((td0 >> 16) | ((td1 & 0xFFFFu) << 16));
                g1[3] = (int)((td1 >> 16) | ((unsigned)BK << 16)); // tile_dim0=32
                g1[4] = (int)(unsigned)BM;                          // tile_dim1=128
                g1[5] = (int)(unsigned)lda;                         // dim0 stride lo
                g1[6] = 0;
                g1[7] = 0;
                v4i g2 = {}, g3 = {};
                v8i g4 = {};
                __builtin_amdgcn_tensor_load_to_lds(g0, g1, g2, g3, g4, 0);
                __builtin_amdgcn_s_wait_tensorcnt((short)0);
            }
        } else {
            // transA fallback: scalar strided gather into the f32 tile
            #pragma unroll
            for (int i = 0; i < 2; ++i) {
                int run = i * 256 + tid, r = run >> 2, koff = (run & 3) * 8;
                int mr = bm + r;
                #pragma unroll
                for (int j = 0; j < 8; ++j) {
                    int kc = k0 + koff + j;
                    float v = 0.0f;
                    if (mr < Mr && kc < Kd)
                        v = Ab[(long long)kc * lda + mr];
                    Af[r][koff + j] = v;
                }
            }
        }
#else
        const bool fastA = (!transA) && ldaAligned && fullK;
        #pragma unroll
        for (int i = 0; i < 2; ++i) {
            int run  = i * 256 + tid;      // 512 runs of 8
            int r    = run >> 2;           // 0..127
            int koff = (run & 3) * 8;
            int mr   = bm + r;
            if (fastA) {
                bf16x8 pk = {};
                if (mr < Mr) {
                    const float* p = Ab + (long long)mr * lda + (k0 + koff);
                    pk = cvt8(*(const fx4*)p, *(const fx4*)(p + 4));
                }
                *(bf16x8*)&As[r][koff] = pk;
            } else {
                #pragma unroll
                for (int j = 0; j < 8; ++j) {
                    int kc = k0 + koff + j;
                    float v = 0.0f;
                    if (mr < Mr && kc < Kd)
                        v = transA ? Ab[(long long)kc * lda + mr]
                                   : Ab[(long long)mr * lda + kc];
                    As[r][koff + j] = (__bf16)v;
                }
            }
        }
#endif

        // ---------- stage B (BK x BN) into Bs[n][k] (bf16, transposed) ----
        if (transB) {
            int n = tid >> 2, koff = (tid & 3) * 8;
            int nc = bn + n;
            if (ldbAligned && fullK) {
                bf16x8 pk = {};
                if (nc < Nc) {
                    const float* p = Bb + (long long)nc * ldb + (k0 + koff);
                    pk = cvt8(*(const fx4*)p, *(const fx4*)(p + 4));
                }
                *(bf16x8*)&Bs[n][koff] = pk;
            } else {
                #pragma unroll
                for (int j = 0; j < 8; ++j) {
                    int kc = k0 + koff + j;
                    float v = 0.0f;
                    if (nc < Nc && kc < Kd)
                        v = Bb[(long long)nc * ldb + kc];
                    Bs[n][koff + j] = (__bf16)v;
                }
            }
        } else {
            int k = tid >> 3, noff = (tid & 7) * 8;
            int kr = k0 + k;
            if (ldbAligned && (bn + noff + 8 <= Nc) && (kr < Kd)) {
                const float* p = Bb + (long long)kr * ldb + (bn + noff);
                fx4 v0 = *(const fx4*)p;
                fx4 v1 = *(const fx4*)(p + 4);
                Bs[noff + 0][k] = (__bf16)v0.x;
                Bs[noff + 1][k] = (__bf16)v0.y;
                Bs[noff + 2][k] = (__bf16)v0.z;
                Bs[noff + 3][k] = (__bf16)v0.w;
                Bs[noff + 4][k] = (__bf16)v1.x;
                Bs[noff + 5][k] = (__bf16)v1.y;
                Bs[noff + 6][k] = (__bf16)v1.z;
                Bs[noff + 7][k] = (__bf16)v1.w;
            } else {
                #pragma unroll
                for (int j = 0; j < 8; ++j) {
                    int nc = bn + noff + j;
                    float v = 0.0f;
                    if (kr < Kd && nc < Nc)
                        v = Bb[(long long)kr * ldb + nc];
                    Bs[noff + j][k] = (__bf16)v;
                }
            }
        }

        if (fullK && k0 + BK < Kd && !transA && bm < Mr)
            __builtin_prefetch(&Ab[(long long)bm * lda + k0 + BK], 0, 1);
        __syncthreads();

        // ---------- fragments ----------
        bf16x16 afrag[2], bfrag[2];
#if USE_TDM
        #pragma unroll
        for (int im = 0; im < 2; ++im) {
            int row = wm * 32 + im * 16 + lr;
            fx4 a0 = *(const fx4*)&Af[row][8 * g];
            fx4 a1 = *(const fx4*)&Af[row][8 * g + 4];
            fx4 a2 = *(const fx4*)&Af[row][16 + 8 * g];
            fx4 a3 = *(const fx4*)&Af[row][16 + 8 * g + 4];
            afrag[im] = __builtin_shufflevector(cvt8(a0, a1), cvt8(a2, a3),
                0, 1, 2, 3, 4, 5, 6, 7, 8, 9, 10, 11, 12, 13, 14, 15);
        }
#else
        #pragma unroll
        for (int im = 0; im < 2; ++im) {
            int row = wm * 32 + im * 16 + lr;
            bf16x8 lo = *(const bf16x8*)&As[row][8 * g];
            bf16x8 hi = *(const bf16x8*)&As[row][16 + 8 * g];
            afrag[im] = __builtin_shufflevector(lo, hi,
                0, 1, 2, 3, 4, 5, 6, 7, 8, 9, 10, 11, 12, 13, 14, 15);
        }
#endif
        #pragma unroll
        for (int in = 0; in < 2; ++in) {
            int col = wn * 32 + in * 16 + lr;
            bf16x8 lo = *(const bf16x8*)&Bs[col][8 * g];
            bf16x8 hi = *(const bf16x8*)&Bs[col][16 + 8 * g];
            bfrag[in] = __builtin_shufflevector(lo, hi,
                0, 1, 2, 3, 4, 5, 6, 7, 8, 9, 10, 11, 12, 13, 14, 15);
        }
        #pragma unroll
        for (int im = 0; im < 2; ++im)
            #pragma unroll
            for (int in = 0; in < 2; ++in)
                acc[im][in] = __builtin_amdgcn_wmma_f32_16x16x32_bf16(
                    false, afrag[im], false, bfrag[in], (short)0,
                    acc[im][in], false, false);
        __syncthreads();
    }

    const float* biasb = bias ? bias + (long long)bat * sBias : nullptr;
    const float* resb  = res  ? res  + (long long)bat * sRes  : nullptr;
    float*       Cb    = C + (long long)bat * sC;

    #pragma unroll
    for (int im = 0; im < 2; ++im)
        #pragma unroll
        for (int in = 0; in < 2; ++in)
            #pragma unroll
            for (int r = 0; r < 8; ++r) {
                int row = bm + wm * 32 + im * 16 + g * 8 + r;
                int col = bn + wn * 32 + in * 16 + lr;
                if (row < Mr && col < Nc) {
                    float v = acc[im][in][r] * scale;
                    if (biasb) v += biasb[col];
                    if (act == 1) v = af_gelu(v);
                    if (resb)  v += resb[(long long)row * ldres + col];
                    Cb[(long long)row * ldc + col] = v;
                }
            }
}

// ---------------------------------------------------------------------
// Fourier PE + concat -> hin (N x 68, cols 66..67 zero)
// ---------------------------------------------------------------------
__global__ void build_hin(const float* __restrict__ pos,
                          const float* __restrict__ feats,
                          float* __restrict__ hin, int Nn) {
    int i = blockIdx.x * blockDim.x + threadIdx.x;
    if (i >= Nn * AF_INPAD) return;
    int n = i / AF_INPAD, c = i % AF_INPAD;
    const float PI = 3.14159265358979323846f;
    float v;
    if (c < 15)      v = feats[n * 15 + c];
    else if (c < 18) v = pos[n * 3 + (c - 15)];
    else if (c < 42) { int j = c - 18; int d = j >> 3, f = j & 7;
                       v = __sinf(pos[n * 3 + d] * (float)(1 << f) * PI); }
    else if (c < 66) { int j = c - 42; int d = j >> 3, f = j & 7;
                       v = __cosf(pos[n * 3 + d] * (float)(1 << f) * PI); }
    else             v = 0.0f;
    hin[i] = v;
}

// Zero-padded weight copy: dst (Rpad x C), src (Rsrc x C)
__global__ void pad_rows(const float* __restrict__ src, float* __restrict__ dst,
                         int Rsrc, int Rpad, int Cc) {
    int i = blockIdx.x * blockDim.x + threadIdx.x;
    if (i >= Rpad * Cc) return;
    int r = i / Cc;
    dst[i] = (r < Rsrc) ? src[i] : 0.0f;
}

// Mean-pool K_AGG nearest points into anchors
__global__ void pool_anchors(const float* __restrict__ loc,
                             const int* __restrict__ idx,
                             float* __restrict__ anc, int Mm, int Dd, int Kk) {
    int i = blockIdx.x * blockDim.x + threadIdx.x;
    if (i >= Mm * Dd) return;
    int m = i / Dd, d = i % Dd;
    float s = 0.0f;
    for (int j = 0; j < Kk; ++j)
        s += loc[(long long)idx[m * Kk + j] * Dd + d];
    anc[i] = s * (1.0f / (float)Kk);
}

// Row LayerNorm (optional broadcast-row add first)
__global__ void layernorm_rows(const float* __restrict__ x,
                               const float* __restrict__ addRow,
                               const float* __restrict__ gg,
                               const float* __restrict__ bb,
                               float* __restrict__ out, int D) {
    __shared__ float r1[128], r2[128];
    long long row = blockIdx.x;
    const float* xr = x + row * D;
    int tid = threadIdx.x;
    float s = 0.0f, s2 = 0.0f;
    for (int c = tid; c < D; c += blockDim.x) {
        float v = xr[c] + (addRow ? addRow[c] : 0.0f);
        s += v; s2 += v * v;
    }
    r1[tid] = s; r2[tid] = s2; __syncthreads();
    for (int st = 64; st > 0; st >>= 1) {
        if (tid < st) { r1[tid] += r1[tid + st]; r2[tid] += r2[tid + st]; }
        __syncthreads();
    }
    float mean = r1[0] / (float)D;
    float var  = r2[0] / (float)D - mean * mean;
    float inv  = rsqrtf(var + 1e-5f);
    for (int c = tid; c < D; c += blockDim.x) {
        float v = xr[c] + (addRow ? addRow[c] : 0.0f);
        out[row * D + c] = (v - mean) * inv * gg[c] + bb[c];
    }
}

// In-place row softmax; scale const or 1/clip(tempPtr[batch], tmin, tmax)
__global__ void softmax_rows(float* __restrict__ x, long long bStride, int cols,
                             const float* __restrict__ tempPtr,
                             float tmin, float tmax, float constScale) {
    __shared__ float red[128];
    int b = blockIdx.y;
    float* row = x + (long long)b * bStride + (long long)blockIdx.x * cols;
    float scale = constScale;
    if (tempPtr) {
        float t = fminf(fmaxf(tempPtr[b], tmin), tmax);
        scale = 1.0f / t;
    }
    int tid = threadIdx.x;
    float m = -3.0e38f;
    for (int c = tid; c < cols; c += blockDim.x) m = fmaxf(m, row[c] * scale);
    red[tid] = m; __syncthreads();
    for (int st = 64; st > 0; st >>= 1) {
        if (tid < st) red[tid] = fmaxf(red[tid], red[tid + st]);
        __syncthreads();
    }
    m = red[0]; __syncthreads();
    float s = 0.0f;
    for (int c = tid; c < cols; c += blockDim.x) {
        float e = __expf(row[c] * scale - m);
        row[c] = e; s += e;
    }
    red[tid] = s; __syncthreads();
    for (int st = 64; st > 0; st >>= 1) {
        if (tid < st) red[tid] += red[tid + st];
        __syncthreads();
    }
    float inv = 1.0f / red[0];
    for (int c = tid; c < cols; c += blockDim.x) row[c] *= inv;
}

// swsum[h*G+g] = sum over m of sw[h, m, g]
__global__ void colsum_over_m(const float* __restrict__ sw,
                              float* __restrict__ out) {
    __shared__ float red[256];
    int hg = blockIdx.x;
    int h = hg / AF_G, g = hg % AF_G;
    const float* base = sw + (long long)h * AF_M * AF_G + g;
    float s = 0.0f;
    for (int m = threadIdx.x; m < AF_M; m += blockDim.x)
        s += base[(long long)m * AF_G];
    red[threadIdx.x] = s; __syncthreads();
    for (int st = 128; st > 0; st >>= 1) {
        if (threadIdx.x < st) red[threadIdx.x] += red[threadIdx.x + st];
        __syncthreads();
    }
    if (threadIdx.x == 0) out[hg] = red[0];
}

__global__ void scale_tok(float* __restrict__ tok,
                          const float* __restrict__ ssum, int total) {
    int i = blockIdx.x * blockDim.x + threadIdx.x;
    if (i >= total) return;
    tok[i] /= (ssum[i / AF_DH] + 1e-5f);
}

// w[n,k] = softmax_k(-clip(exp(log_temp),0.5,200)*qd[n,k])
__global__ void readout_weights(const float* __restrict__ qdist,
                                const float* __restrict__ logTemp,
                                float* __restrict__ w, int Nn) {
    int n = blockIdx.x * blockDim.x + threadIdx.x;
    if (n >= Nn) return;
    float t = fminf(fmaxf(__expf(logTemp[0]), 0.5f), 200.0f);
    float mx = -3.0e38f;
    float l[AF_KQ];
    #pragma unroll
    for (int k = 0; k < AF_KQ; ++k) {
        l[k] = -t * qdist[n * AF_KQ + k];
        mx = fmaxf(mx, l[k]);
    }
    float s = 0.0f;
    #pragma unroll
    for (int k = 0; k < AF_KQ; ++k) { l[k] = __expf(l[k] - mx); s += l[k]; }
    float inv = 1.0f / s;
    #pragma unroll
    for (int k = 0; k < AF_KQ; ++k) w[n * AF_KQ + k] = l[k] * inv;
}

// comb[n,0:256] = sum_k w[n,k]*x[qidx[n,k],:]; comb[n,256:384] = loc[n,:]
__global__ void pool_points(const float* __restrict__ w,
                            const int* __restrict__ qidx,
                            const float* __restrict__ x,
                            const float* __restrict__ loc,
                            float* __restrict__ comb, int Nn) {
    int i = blockIdx.x * blockDim.x + threadIdx.x;
    int total = Nn * (AF_DANC + AF_DLOC);
    if (i >= total) return;
    int n = i / (AF_DANC + AF_DLOC), d = i % (AF_DANC + AF_DLOC);
    float v;
    if (d < AF_DANC) {
        v = 0.0f;
        #pragma unroll
        for (int k = 0; k < AF_KQ; ++k)
            v += w[n * AF_KQ + k] * x[(long long)qidx[n * AF_KQ + k] * AF_DANC + d];
    } else {
        v = loc[(long long)n * AF_DLOC + (d - AF_DANC)];
    }
    comb[i] = v;
}

// =====================================================================
// Host side
// =====================================================================
static inline void launch_gemm(hipStream_t s,
        const float* A, int lda, long long sA, int tA,
        const float* B, int ldb, long long sB, int tB,
        const float* bias, long long sBias,
        const float* res, long long sRes, int ldres,
        float* C, int ldc, long long sC,
        int Mr, int Nc, int Kd, int batch, float scale, int act) {
    dim3 grid((Mr + BM - 1) / BM, (Nc + BN - 1) / BN, batch);
    wmma_gemm<<<grid, 256, 0, s>>>(A, lda, sA, tA, B, ldb, sB, tB,
                                   bias, sBias, res, sRes, ldres,
                                   C, ldc, sC, Mr, Nc, Kd, scale, act);
}

static inline dim3 g1(long long n, int bs) { return dim3((unsigned)((n + bs - 1) / bs)); }

// Input index assumption: harness flattens setup_inputs() pytree in jax tree
// order (dict keys sorted alphabetically, depth-first).
enum {
    I_FEATS = 0, I_NN_IDX_AGG = 1,
    I_AENC_W = 2, I_AENC_B = 3,
    I_B_W1 = 4, I_B_W2 = 5, I_B_WFX = 6, I_B_WOUT = 7, I_B_WSLICE = 8,
    I_B_WX = 9, I_B_WK = 10, I_B_WQ = 11, I_B_WV = 12,
    I_B_B1 = 13, I_B_B2 = 14, I_B_BFX = 15, I_B_BOUT = 16, I_B_BSLICE = 17,
    I_B_BX = 18, I_B_LN1B = 19, I_B_LN1G = 20, I_B_LN2B = 21, I_B_LN2G = 22,
    I_B_TEMP = 23,
    I_ENC_W1 = 24, I_ENC_W2 = 25, I_ENC_B1 = 26, I_ENC_B2 = 27,
    I_H_W1 = 28, I_H_W2 = 29, I_H_W3 = 30, I_H_B1 = 31, I_H_B2 = 32,
    I_H_B3 = 33, I_H_LNB = 34, I_H_LNG = 35,
    I_LOG_TEMP = 36,
    I_RO_W1 = 37, I_RO_W2 = 38, I_RO_B1 = 39, I_RO_B2 = 40,
    I_RO_LNB = 41, I_RO_LNG = 42,
    I_STEP_B = 43,
    I_POS = 44, I_QDIST = 45, I_QIDX = 46
};

extern "C" void kernel_launch(void* const* d_in, const int* in_sizes, int n_in,
                              void* d_out, int out_size, void* d_ws, size_t ws_size,
                              hipStream_t stream) {
    (void)in_sizes; (void)n_in; (void)out_size; (void)ws_size;

    auto F = [&](int i) { return (const float*)d_in[i]; };
    auto I = [&](int i) { return (const int*)d_in[i]; };

    const float* pos    = F(I_POS);
    const float* feats  = F(I_FEATS);
    const float* qdist  = F(I_QDIST);
    const int*   nnagg  = I(I_NN_IDX_AGG);
    const int*   qidx   = I(I_QIDX);

    // ---- workspace (linear allocator, f32) ----
    float* W = (float*)d_ws;
    size_t o = 0;
    auto alloc = [&](size_t n) { float* p = W + o; o += n; return p; };

    float* hin   = alloc((size_t)AF_N * AF_INPAD);
    float* eW1p  = alloc((size_t)AF_INPAD * AF_DLOC);
    float* h1    = alloc((size_t)AF_N * AF_DLOC);
    float* loc   = alloc((size_t)AF_N * AF_DLOC);
    float* anc0  = alloc((size_t)AF_M * AF_DLOC);
    float* x     = alloc((size_t)AF_M * AF_DANC);
    float* hb    = alloc((size_t)AF_M * AF_DANC);
    float* fx    = alloc((size_t)AF_M * AF_DANC);
    float* xm    = alloc((size_t)AF_M * AF_DANC);
    float* sw    = alloc((size_t)AF_H * AF_M * AF_G);
    float* swsum = alloc((size_t)AF_H * AF_G);
    float* tok   = alloc((size_t)AF_H * AF_G * AF_DH);
    float* qb    = alloc((size_t)AF_H * AF_G * AF_DH);
    float* kb    = alloc((size_t)AF_H * AF_G * AF_DH);
    float* vb    = alloc((size_t)AF_H * AF_G * AF_DH);
    float* attn  = alloc((size_t)AF_H * AF_G * AF_G);
    float* outsl = alloc((size_t)AF_H * AF_G * AF_DH);
    float* oxb   = alloc((size_t)AF_M * AF_DANC);
    float* ffn1  = alloc((size_t)AF_M * 4 * AF_DANC);
    float* wbuf  = alloc((size_t)AF_N * AF_KQ);
    float* comb  = alloc((size_t)AF_N * (AF_DANC + AF_DLOC));
    float* combn = alloc((size_t)AF_N * (AF_DANC + AF_DLOC));
    float* roh1  = alloc((size_t)AF_N * 2 * AF_DANC);
    float* roh   = alloc((size_t)AF_N * AF_DANC);
    float* z1    = alloc((size_t)AF_N * AF_DANC);
    float* z2    = alloc((size_t)AF_N * AF_DANC);

    float* out = (float*)d_out;   // (N, T_OUT, 3), f32

    // ================= encoder =================
    build_hin<<<g1((long long)AF_N * AF_INPAD, 256), 256, 0, stream>>>(pos, feats, hin, AF_N);
    pad_rows<<<g1((long long)AF_INPAD * AF_DLOC, 256), 256, 0, stream>>>(
        F(I_ENC_W1), eW1p, AF_INDIM, AF_INPAD, AF_DLOC);

    launch_gemm(stream, hin, AF_INPAD, 0, 0, eW1p, AF_DLOC, 0, 0,
                F(I_ENC_B1), 0, nullptr, 0, 0,
                h1, AF_DLOC, 0, AF_N, AF_DLOC, AF_INPAD, 1, 1.0f, 1);
    launch_gemm(stream, h1, AF_DLOC, 0, 0, F(I_ENC_W2), AF_DLOC, 0, 0,
                F(I_ENC_B2), 0, nullptr, 0, 0,
                loc, AF_DLOC, 0, AF_N, AF_DLOC, AF_DLOC, 1, 1.0f, 0);

    pool_anchors<<<g1((long long)AF_M * AF_DLOC, 256), 256, 0, stream>>>(
        loc, nnagg, anc0, AF_M, AF_DLOC, AF_KAGG);
    launch_gemm(stream, anc0, AF_DLOC, 0, 0, F(I_AENC_W), AF_DANC, 0, 0,
                F(I_AENC_B), 0, nullptr, 0, 0,
                x, AF_DANC, 0, AF_M, AF_DANC, AF_DLOC, 1, 1.0f, 0);

    // ================= 12 anchor blocks =================
    for (int l = 0; l < AF_L; ++l) {
        const float* Wfx = F(I_B_WFX) + (size_t)l * AF_DANC * AF_DANC;
        const float* Wx_ = F(I_B_WX)  + (size_t)l * AF_DANC * AF_DANC;
        const float* Wsl = F(I_B_WSLICE) + (size_t)l * AF_DH * AF_G;
        const float* Wq_ = F(I_B_WQ) + (size_t)l * AF_DH * AF_DH;
        const float* Wk_ = F(I_B_WK) + (size_t)l * AF_DH * AF_DH;
        const float* Wv_ = F(I_B_WV) + (size_t)l * AF_DH * AF_DH;
        const float* Wo  = F(I_B_WOUT) + (size_t)l * AF_DANC * AF_DANC;
        const float* W1_ = F(I_B_W1) + (size_t)l * AF_DANC * 4 * AF_DANC;
        const float* W2_ = F(I_B_W2) + (size_t)l * 4 * AF_DANC * AF_DANC;

        layernorm_rows<<<AF_M, 128, 0, stream>>>(
            x, nullptr, F(I_B_LN1G) + l * AF_DANC, F(I_B_LN1B) + l * AF_DANC, hb, AF_DANC);

        launch_gemm(stream, hb, AF_DANC, 0, 0, Wfx, AF_DANC, 0, 0,
                    F(I_B_BFX) + l * AF_DANC, 0, nullptr, 0, 0,
                    fx, AF_DANC, 0, AF_M, AF_DANC, AF_DANC, 1, 1.0f, 0);
        launch_gemm(stream, hb, AF_DANC, 0, 0, Wx_, AF_DANC, 0, 0,
                    F(I_B_BX) + l * AF_DANC, 0, nullptr, 0, 0,
                    xm, AF_DANC, 0, AF_M, AF_DANC, AF_DANC, 1, 1.0f, 0);

        launch_gemm(stream, xm + 0, AF_DANC, AF_DH, 0,
                    Wsl, AF_G, 0, 0,
                    F(I_B_BSLICE) + l * AF_G, 0, nullptr, 0, 0,
                    sw, AF_G, (long long)AF_M * AF_G,
                    AF_M, AF_G, AF_DH, AF_H, 1.0f, 0);
        softmax_rows<<<dim3(AF_M, AF_H), 128, 0, stream>>>(
            sw, (long long)AF_M * AF_G, AF_G, F(I_B_TEMP) + l * AF_H, 0.1f, 5.0f, 1.0f);

        colsum_over_m<<<AF_H * AF_G, 256, 0, stream>>>(sw, swsum);
        launch_gemm(stream, sw, AF_G, (long long)AF_M * AF_G, 1,   // transA
                    fx + 0, AF_DANC, AF_DH, 0,
                    nullptr, 0, nullptr, 0, 0,
                    tok, AF_DH, (long long)AF_G * AF_DH,
                    AF_G, AF_DH, AF_M, AF_H, 1.0f, 0);
        scale_tok<<<g1((long long)AF_H * AF_G * AF_DH, 256), 256, 0, stream>>>(
            tok, swsum, AF_H * AF_G * AF_DH);

        launch_gemm(stream, tok, AF_DH, 0, 0, Wq_, AF_DH, 0, 0,
                    nullptr, 0, nullptr, 0, 0,
                    qb, AF_DH, 0, AF_H * AF_G, AF_DH, AF_DH, 1, 1.0f, 0);
        launch_gemm(stream, tok, AF_DH, 0, 0, Wk_, AF_DH, 0, 0,
                    nullptr, 0, nullptr, 0, 0,
                    kb, AF_DH, 0, AF_H * AF_G, AF_DH, AF_DH, 1, 1.0f, 0);
        launch_gemm(stream, tok, AF_DH, 0, 0, Wv_, AF_DH, 0, 0,
                    nullptr, 0, nullptr, 0, 0,
                    vb, AF_DH, 0, AF_H * AF_G, AF_DH, AF_DH, 1, 1.0f, 0);

        launch_gemm(stream, qb, AF_DH, (long long)AF_G * AF_DH, 0,
                    kb, AF_DH, (long long)AF_G * AF_DH, 1,        // transB
                    nullptr, 0, nullptr, 0, 0,
                    attn, AF_G, (long long)AF_G * AF_G,
                    AF_G, AF_G, AF_DH, AF_H, 0.25f, 0);
        softmax_rows<<<dim3(AF_G, AF_H), 128, 0, stream>>>(
            attn, (long long)AF_G * AF_G, AF_G, nullptr, 0.0f, 0.0f, 1.0f);

        launch_gemm(stream, attn, AF_G, (long long)AF_G * AF_G, 0,
                    vb, AF_DH, (long long)AF_G * AF_DH, 0,
                    nullptr, 0, nullptr, 0, 0,
                    outsl, AF_DH, (long long)AF_G * AF_DH,
                    AF_G, AF_DH, AF_G, AF_H, 1.0f, 0);

        launch_gemm(stream, sw, AF_G, (long long)AF_M * AF_G, 0,
                    outsl, AF_DH, (long long)AF_G * AF_DH, 0,
                    nullptr, 0, nullptr, 0, 0,
                    oxb + 0, AF_DANC, AF_DH,
                    AF_M, AF_DH, AF_G, AF_H, 1.0f, 0);

        launch_gemm(stream, oxb, AF_DANC, 0, 0, Wo, AF_DANC, 0, 0,
                    F(I_B_BOUT) + l * AF_DANC, 0, x, 0, AF_DANC,
                    x, AF_DANC, 0, AF_M, AF_DANC, AF_DANC, 1, 1.0f, 0);

        layernorm_rows<<<AF_M, 128, 0, stream>>>(
            x, nullptr, F(I_B_LN2G) + l * AF_DANC, F(I_B_LN2B) + l * AF_DANC, hb, AF_DANC);
        launch_gemm(stream, hb, AF_DANC, 0, 0, W1_, 4 * AF_DANC, 0, 0,
                    F(I_B_B1) + l * 4 * AF_DANC, 0, nullptr, 0, 0,
                    ffn1, 4 * AF_DANC, 0, AF_M, 4 * AF_DANC, AF_DANC, 1, 1.0f, 1);
        launch_gemm(stream, ffn1, 4 * AF_DANC, 0, 0, W2_, AF_DANC, 0, 0,
                    F(I_B_B2) + l * AF_DANC, 0, x, 0, AF_DANC,
                    x, AF_DANC, 0, AF_M, AF_DANC, 4 * AF_DANC, 1, 1.0f, 0);
    }

    // ================= anchor readout =================
    readout_weights<<<g1(AF_N, 256), 256, 0, stream>>>(qdist, F(I_LOG_TEMP), wbuf, AF_N);
    pool_points<<<g1((long long)AF_N * (AF_DANC + AF_DLOC), 256), 256, 0, stream>>>(
        wbuf, qidx, x, loc, comb, AF_N);
    layernorm_rows<<<AF_N, 128, 0, stream>>>(
        comb, nullptr, F(I_RO_LNG), F(I_RO_LNB), combn, AF_DANC + AF_DLOC);
    launch_gemm(stream, combn, AF_DANC + AF_DLOC, 0, 0, F(I_RO_W1), 2 * AF_DANC, 0, 0,
                F(I_RO_B1), 0, nullptr, 0, 0,
                roh1, 2 * AF_DANC, 0, AF_N, 2 * AF_DANC, AF_DANC + AF_DLOC, 1, 1.0f, 1);
    launch_gemm(stream, roh1, 2 * AF_DANC, 0, 0, F(I_RO_W2), AF_DANC, 0, 0,
                F(I_RO_B2), 0, nullptr, 0, 0,
                roh, AF_DANC, 0, AF_N, AF_DANC, 2 * AF_DANC, 1, 1.0f, 0);

    // ================= per-step heads =================
    for (int t = 0; t < AF_TOUT; ++t) {
        layernorm_rows<<<AF_N, 128, 0, stream>>>(
            roh, F(I_STEP_B) + t * AF_DANC,
            F(I_H_LNG) + t * AF_DANC, F(I_H_LNB) + t * AF_DANC, z1, AF_DANC);
        launch_gemm(stream, z1, AF_DANC, 0, 0,
                    F(I_H_W1) + (size_t)t * AF_DANC * AF_DANC, AF_DANC, 0, 0,
                    F(I_H_B1) + t * AF_DANC, 0, nullptr, 0, 0,
                    z2, AF_DANC, 0, AF_N, AF_DANC, AF_DANC, 1, 1.0f, 1);
        launch_gemm(stream, z2, AF_DANC, 0, 0,
                    F(I_H_W2) + (size_t)t * AF_DANC * (AF_DANC / 2), AF_DANC / 2, 0, 0,
                    F(I_H_B2) + t * (AF_DANC / 2), 0, nullptr, 0, 0,
                    z1, AF_DANC / 2, 0, AF_N, AF_DANC / 2, AF_DANC, 1, 1.0f, 1);
        launch_gemm(stream, z1, AF_DANC / 2, 0, 0,
                    F(I_H_W3) + (size_t)t * (AF_DANC / 2) * 3, 3, 0, 0,
                    F(I_H_B3) + t * 3, 0, nullptr, 0, 0,
                    out + t * 3, AF_TOUT * 3, 0, AF_N, 3, AF_DANC / 2, 1, 1.0f, 0);
    }
}
